// BaselineDynamics_20572893348359
// MI455X (gfx1250) — matrix operand order repaired and checked by reading, not compiled, not run on hardware
//
#include <hip/hip_runtime.h>

#define LATENT 18
#define HID    32
#define BB     2048
#define TT     512
#define NT_G   8          // 8 N-tiles of 16 => 128 gate columns

typedef __attribute__((ext_vector_type(16))) __bf16 v16bf;
typedef __attribute__((ext_vector_type(8)))  __bf16 v8bf;
typedef __attribute__((ext_vector_type(8)))  float  v8f;

#if __has_builtin(__builtin_amdgcn_tanhf)
#define HW_TANH(x) __builtin_amdgcn_tanhf(x)
#define HAS_HW_TANH 1
#elif __has_builtin(__builtin_amdgcn_tanh_f32)
#define HW_TANH(x) __builtin_amdgcn_tanh_f32(x)
#define HAS_HW_TANH 1
#else
#define HAS_HW_TANH 0
#endif

__device__ __forceinline__ float ftanh(float x) {
#if HAS_HW_TANH
    return HW_TANH(x);
#else
    return 1.0f - 2.0f * __builtin_amdgcn_rcpf(1.0f + __expf(2.0f * x));
#endif
}
__device__ __forceinline__ float fsig(float x) {
#if HAS_HW_TANH
    return 0.5f + 0.5f * HW_TANH(0.5f * x);   // sigmoid via native v_tanh_f32
#else
    return __builtin_amdgcn_rcpf(1.0f + __expf(-x));
#endif
}

// One wave (32 lanes) owns a 16-row batch tile for all T timesteps.
// Gates: [z_t | 1 | h] x [Wih ; bias ; Whh] (K=64 bf16) via 16x v_wmma_f32_16x16x32_bf16
// starting from C = inline 0 (bias folded into K=18 row). Two passes (i,g then f,o).
// Output proj: h (K=32) x Wout^T + hoisted b_out splat via 2 more WMMA per step.
__global__ __launch_bounds__(32)
void lstm_fused_wmma(const float* __restrict__ z,
                     const float* __restrict__ Wih,   // [128][18]
                     const float* __restrict__ Whh,   // [128][32]
                     const float* __restrict__ bih,   // [128]
                     const float* __restrict__ bhh,   // [128]
                     const float* __restrict__ Wout,  // [18][32]
                     const float* __restrict__ bout,  // [18]
                     float* __restrict__ out)
{
    __shared__ __bf16 hbuf[16 * HID];   // 1 KB h tile for C-layout -> A-layout shuffle

    const int lane = threadIdx.x & 31;
    const int l15  = lane & 15;
    const int lh   = lane >> 4;         // lane half: 0 (lanes 0-15) / 1 (lanes 16-31)
    const int b0   = blockIdx.x * 16;   // batch tile base

    // ---------------- resident weights -----------------------------------
    // B-matrix 32x16 bf16 layout: element i of v16bf holds K = (i<8 ? 8*lh + i : 16 + 8*lh + i - 8), N = l15.
    v16bf Bw0[NT_G];                    // z-part weights; row K=18 carries b_ih+b_hh
    v16bf Bw1[NT_G];                    // h-part weights (W_hh)
    #pragma unroll
    for (int nt = 0; nt < NT_G; ++nt) {
        const int N = nt * 16 + l15;
        const float bias = bih[N] + bhh[N];
        v16bf b0v, b1v;
        #pragma unroll
        for (int i = 0; i < 16; ++i) {
            const int K  = (i < 8) ? (8 * lh + i) : (16 + 8 * lh + (i - 8));  // 0..31
            const int Kc = (K < LATENT) ? K : 0;
            float w0 = (K < LATENT) ? Wih[N * LATENT + Kc]
                                    : ((K == LATENT) ? bias : 0.0f);
            float w1 = Whh[N * HID + K];
            b0v[i] = (__bf16)w0;
            b1v[i] = (__bf16)w1;
        }
        Bw0[nt] = b0v;
        Bw1[nt] = b1v;
    }
    v16bf Bo[2];                        // Wout^T tiles (N = 0..31, cols >=18 zero)
    #pragma unroll
    for (int n2 = 0; n2 < 2; ++n2) {
        const int N = n2 * 16 + l15;
        v16bf bv;
        #pragma unroll
        for (int i = 0; i < 16; ++i) {
            const int K = (i < 8) ? (8 * lh + i) : (16 + 8 * lh + (i - 8));
            float w = (N < LATENT) ? Wout[(N < LATENT ? N : 0) * HID + K] : 0.0f;
            bv[i] = (__bf16)w;
        }
        Bo[n2] = bv;
    }
    // hoisted b_out splat: loop-invariant WMMA C operand (16 VGPRs resident)
    v8f obv[2];
    #pragma unroll
    for (int n2 = 0; n2 < 2; ++n2) {
        const int N = n2 * 16 + l15;
        const float ob = (N < LATENT) ? bout[N] : 0.0f;
        #pragma unroll
        for (int r = 0; r < 8; ++r) obv[n2][r] = ob;
    }
    // bias-activation "1" that lands at A-element K=18 (element 10, lanes 0-15 only)
    const __bf16 one18 = lh ? (__bf16)0.0f : (__bf16)1.0f;

    // ---------------- state ----------------------------------------------
    float cst[2][8];                    // cell state, C-layout fp32
    float hfv[2][8];                    // last h, C-layout fp32
    #pragma unroll
    for (int ht = 0; ht < 2; ++ht)
        #pragma unroll
        for (int r = 0; r < 8; ++r) { cst[ht][r] = 0.0f; hfv[ht][r] = 0.0f; }

    v16bf a_h;                          // h_{t-1} in A-matrix bf16 layout
    #pragma unroll
    for (int i = 0; i < 16; ++i) a_h[i] = (__bf16)0.0f;

    float* zp = out;                                          // z_pred [B][T][18]
    float* hf = out + (size_t)BB * TT * LATENT;               // h_f    [B][32]
    float* cf = hf + (size_t)BB * HID;                        // c_f    [B][32]

    // one gate N-tile: inline-0 C -> + [z|1]*[Wih;bias] -> + h*Whh (f32 accumulate)
    auto gate_tile = [&](int nt, const v16bf& a_z) -> v8f {
        v8f c = {0.0f, 0.0f, 0.0f, 0.0f, 0.0f, 0.0f, 0.0f, 0.0f};
        c = __builtin_amdgcn_wmma_f32_16x16x32_bf16(false, a_z, false, Bw0[nt],
                                                    (short)0, c, false, false);
        c = __builtin_amdgcn_wmma_f32_16x16x32_bf16(false, a_h, false, Bw1[nt],
                                                    (short)0, c, false, false);
        return c;
    };

    #pragma unroll 1
    for (int t = 0; t < TT; ++t) {
        // ---- A-fragment from z_t (K padded 18->32): lane holds K in {lh*8..+7, 16+lh*8..}
        const float* zrow = z + ((size_t)(b0 + l15) * TT + t) * LATENT;
        const float* zr   = zrow + lh * 8;
        float2 p0 = *(const float2*)(zr + 0);
        float2 p1 = *(const float2*)(zr + 2);
        float2 p2 = *(const float2*)(zr + 4);
        float2 p3 = *(const float2*)(zr + 6);
        // unconditional tail loads (always in-bounds), then lane-select
        float e16 = zrow[16];
        float e17 = zrow[17];
        e16 = lh ? 0.0f : e16;
        e17 = lh ? 0.0f : e17;
        v16bf a_z;
        a_z[0]  = (__bf16)p0.x;  a_z[1] = (__bf16)p0.y;
        a_z[2]  = (__bf16)p1.x;  a_z[3] = (__bf16)p1.y;
        a_z[4]  = (__bf16)p2.x;  a_z[5] = (__bf16)p2.y;
        a_z[6]  = (__bf16)p3.x;  a_z[7] = (__bf16)p3.y;
        a_z[8]  = (__bf16)e16;   a_z[9] = (__bf16)e17;
        a_z[10] = one18;                       // K=18: bias multiplier (1.0 on lanes 0-15)
        #pragma unroll
        for (int i = 11; i < 16; ++i) a_z[i] = (__bf16)0.0f;

        // ---- pass 1: i (tiles 0,1) and g (tiles 4,5) -> ig = sigmoid(i)*tanh(g)
        v8f ai0 = gate_tile(0, a_z);
        v8f ai1 = gate_tile(1, a_z);
        v8f ag0 = gate_tile(4, a_z);
        v8f ag1 = gate_tile(5, a_z);
        float ig[2][8];
        #pragma unroll
        for (int r = 0; r < 8; ++r) {
            ig[0][r] = fsig(ai0[r]) * ftanh(ag0[r]);
            ig[1][r] = fsig(ai1[r]) * ftanh(ag1[r]);
        }

        // ---- pass 2: f (tiles 2,3) and o (tiles 6,7) -> c,h update; h -> LDS (bf16)
        v8f af0 = gate_tile(2, a_z);
        v8f af1 = gate_tile(3, a_z);
        v8f ao0 = gate_tile(6, a_z);
        v8f ao1 = gate_tile(7, a_z);
        #pragma unroll
        for (int r = 0; r < 8; ++r) {
            float cc0 = fsig(af0[r]) * cst[0][r] + ig[0][r];
            float cc1 = fsig(af1[r]) * cst[1][r] + ig[1][r];
            cst[0][r] = cc0;
            cst[1][r] = cc1;
            float hh0 = fsig(ao0[r]) * ftanh(cc0);
            float hh1 = fsig(ao1[r]) * ftanh(cc1);
            hfv[0][r] = hh0;
            hfv[1][r] = hh1;
            hbuf[(r + 8 * lh) * HID + l15     ] = (__bf16)hh0;   // row-major h tile
            hbuf[(r + 8 * lh) * HID + l15 + 16] = (__bf16)hh1;
        }

        // ---- rebuild A-layout h fragment: two ds_load_b128 per lane (in-order DS, 1 wave)
        v8bf hlo = *(const v8bf*)&hbuf[l15 * HID + lh * 8];
        v8bf hhi = *(const v8bf*)&hbuf[l15 * HID + 16 + lh * 8];
        #pragma unroll
        for (int i = 0; i < 8; ++i) { a_h[i] = hlo[i]; a_h[8 + i] = hhi[i]; }

        // ---- output projection: z_pred[:,t,:] = h @ Wout^T + b_out (2 WMMA)
        #pragma unroll
        for (int n2 = 0; n2 < 2; ++n2) {
            v8f oc = __builtin_amdgcn_wmma_f32_16x16x32_bf16(false, a_h, false, Bo[n2],
                                                             (short)0, obv[n2], false, false);
            const int N = n2 * 16 + l15;
            if (N < LATENT) {
                #pragma unroll
                for (int r = 0; r < 8; ++r) {
                    const int row = b0 + r + 8 * lh;
                    zp[((size_t)row * TT + t) * LATENT + N] = oc[r];
                }
            }
        }
    }

    // ---- final hidden / cell state: [1][B][32] each, fp32 C-layout scatter
    #pragma unroll
    for (int ht = 0; ht < 2; ++ht) {
        #pragma unroll
        for (int r = 0; r < 8; ++r) {
            const int row = b0 + r + 8 * lh;
            const int col = l15 + 16 * ht;
            hf[(size_t)row * HID + col] = hfv[ht][r];
            cf[(size_t)row * HID + col] = cst[ht][r];
        }
    }
}

extern "C" void kernel_launch(void* const* d_in, const int* in_sizes, int n_in,
                              void* d_out, int out_size, void* d_ws, size_t ws_size,
                              hipStream_t stream) {
    (void)in_sizes; (void)n_in; (void)d_ws; (void)ws_size; (void)out_size;
    const float* z    = (const float*)d_in[0];
    const float* Wih  = (const float*)d_in[1];
    const float* Whh  = (const float*)d_in[2];
    const float* bih  = (const float*)d_in[3];
    const float* bhh  = (const float*)d_in[4];
    const float* Wout = (const float*)d_in[5];
    const float* bout = (const float*)d_in[6];
    float* out = (float*)d_out;

    dim3 grid(BB / 16);   // 128 batch tiles, one wave each
    dim3 block(32);
    lstm_fused_wmma<<<grid, block, 0, stream>>>(z, Wih, Whh, bih, bhh, Wout, bout, out);
}
